// Entropy_encode_81552839016927
// MI455X (gfx1250) — matrix operand order repaired and checked by reading, not compiled, not run on hardware
//
#include <hip/hip_runtime.h>
#include <math.h>

// ---------------------------------------------------------------------------
// CDNA5 / gfx1250 fused attention block, WMMA f16->f32 everywhere.
//   B=256, S=512, D_IN=256, D_ATT=512.
// Pipeline:
//   0) pre-pass: x -> f16 (xh), weights -> f16 transposed (Wt*: [n][k])
//   1) LDS-free WMMA GEMMs: Q/K/V = xh @ W + b + pe   (operands consumed
//      directly from global in WMMA fragment layout; weights are L2-resident)
//   2) fused flash attention + V residual + LayerNorm (f32 + f16 outputs)
//   3) LDS-free WMMA GEMM: out = LN @ Wf + bf + LN    (f32 residual)
// ---------------------------------------------------------------------------

#define BATCH 256
#define SEQ   512
#define DIN   256
#define DATT  512

typedef __attribute__((ext_vector_type(16))) _Float16 v16h;
typedef __attribute__((ext_vector_type(8)))  float    v8f;

union Acc { v8f v; float f[8]; };

__device__ __forceinline__ v8f wmma32(v16h a, v16h b, v8f c) {
  // D = A(16x32 f16) * B(32x16 f16) + C(16x16 f32)
  return __builtin_amdgcn_wmma_f32_16x16x32_f16(
      false, a, false, b, (short)0, c, false, false);
}

// A fragment (16x32 f16): lane L holds row (L&15); halves 0..7 -> K=hi*8..,
// halves 8..15 -> K=16+hi*8..  (ISA 7.12.2).  Two 16B loads.
__device__ __forceinline__ v16h frag_a(const _Float16* p, size_t stride,
                                       int ln, int hi) {
  union { v16h v; uint4 q[2]; } u;
  const _Float16* r = p + (size_t)ln * stride + hi * 8;
  u.q[0] = *(const uint4*)(r);
  u.q[1] = *(const uint4*)(r + 16);
  return u.v;
}

// B fragment (32x16 f16): lane L holds column (L&15); halves 0..7 -> K=hi*8.., halves 8..15 -> K=16+hi*8..
// Column must be K-contiguous at p + col*stride.
__device__ __forceinline__ v16h frag_b(const _Float16* p, size_t stride,
                                       int ln, int hi) {
  union { v16h v; uint4 q[2]; } u;
  const _Float16* r = p + (size_t)ln * stride + hi * 8;
  u.q[0] = *(const uint4*)(r);
  u.q[1] = *(const uint4*)(r + 16);
  return u.v;
}

__device__ __forceinline__ float redmax16(float v) {
  v = fmaxf(v, __shfl_xor(v, 1, 32));
  v = fmaxf(v, __shfl_xor(v, 2, 32));
  v = fmaxf(v, __shfl_xor(v, 4, 32));
  v = fmaxf(v, __shfl_xor(v, 8, 32));
  return v;
}
__device__ __forceinline__ float redsum16(float v) {
  v += __shfl_xor(v, 1, 32);
  v += __shfl_xor(v, 2, 32);
  v += __shfl_xor(v, 4, 32);
  v += __shfl_xor(v, 8, 32);
  return v;
}

// ---------------------------------------------------------------------------
// Pre-pass: f32 -> f16 bulk convert (b128 in / b128 out).
// ---------------------------------------------------------------------------
__global__ __launch_bounds__(256) void cvt_f16_kernel(
    const float* __restrict__ src, _Float16* __restrict__ dst, long n8) {
  const long i = (long)blockIdx.x * blockDim.x + threadIdx.x;
  if (i >= n8) return;
  const float* s = src + i * 8;
  float4 a = *(const float4*)(s);
  float4 b = *(const float4*)(s + 4);
  union { uint4 q; _Float16 h[8]; } o;
  o.h[0] = (_Float16)a.x; o.h[1] = (_Float16)a.y;
  o.h[2] = (_Float16)a.z; o.h[3] = (_Float16)a.w;
  o.h[4] = (_Float16)b.x; o.h[5] = (_Float16)b.y;
  o.h[6] = (_Float16)b.z; o.h[7] = (_Float16)b.w;
  *(uint4*)(dst + i * 8) = o.q;
}

// Pre-pass: weight transpose+convert: W[k][512] f32 -> Wt[n][kdim] f16.
__global__ __launch_bounds__(256) void cvt_wt_kernel(
    const float* __restrict__ W, _Float16* __restrict__ Wt, int kdim) {
  const int n = blockIdx.x * 32 + (threadIdx.x & 31);
  const int k = blockIdx.y * 8 + (threadIdx.x >> 5);
  if (k < kdim) Wt[(size_t)n * kdim + k] = (_Float16)W[(size_t)k * 512 + n];
}

// ---------------------------------------------------------------------------
// LDS-free WMMA GEMM:  C[M,512] = Ah[M,kdim](f16) * Wt[512,kdim](f16, n-major)
//   + bias[col] (+ pe[row%512, col]) (+ resid[row,col]); out f16 and/or f32.
// Block 256 thr (8 waves, 2x4), tile 128x128, BK=32; all operand loads are
// global_load_b128 directly in fragment layout (weights/activations L2-hot).
// ---------------------------------------------------------------------------
__global__ __launch_bounds__(256) void gemm_f16g_kernel(
    const _Float16* __restrict__ Ah, int kdim,
    const _Float16* __restrict__ Wt, const float* __restrict__ bias,
    const float* __restrict__ pe, const float* __restrict__ resid,
    _Float16* __restrict__ outH, float* __restrict__ outF) {
  const int tid = threadIdx.x;
  const int lane = tid & 31;
  const int w = tid >> 5;
  const int ln = lane & 15, hi = lane >> 4;
  const int wm = w & 1, wn = w >> 1;           // 2 x 4 wave grid
  const size_t rowBase = (size_t)blockIdx.y * 128;
  const int n0 = blockIdx.x * 128;

  Acc c[4][2];
#pragma unroll
  for (int mt = 0; mt < 4; ++mt)
#pragma unroll
    for (int nt = 0; nt < 2; ++nt)
#pragma unroll
      for (int r = 0; r < 8; ++r) c[mt][nt].f[r] = 0.0f;

  const _Float16* Abase = Ah + (rowBase + wm * 64) * (size_t)kdim;
  const _Float16* Bbase = Wt + (size_t)(n0 + wn * 32) * kdim;

  for (int k0 = 0; k0 < kdim; k0 += 32) {
    v16h bf0 = frag_b(Bbase + k0, kdim, ln, hi);
    v16h bf1 = frag_b(Bbase + (size_t)16 * kdim + k0, kdim, ln, hi);
#pragma unroll
    for (int mt = 0; mt < 4; ++mt) {
      v16h af = frag_a(Abase + (size_t)(mt * 16) * kdim + k0, kdim, ln, hi);
      c[mt][0].v = wmma32(af, bf0, c[mt][0].v);
      c[mt][1].v = wmma32(af, bf1, c[mt][1].v);
    }
  }

#pragma unroll
  for (int mt = 0; mt < 4; ++mt) {
#pragma unroll
    for (int nt = 0; nt < 2; ++nt) {
      const int col = n0 + wn * 32 + nt * 16 + ln;
      const float bc = bias[col];
#pragma unroll
      for (int r = 0; r < 8; ++r) {
        const int rl = wm * 64 + mt * 16 + r + 8 * hi;
        const size_t grow = rowBase + rl;
        float v = c[mt][nt].f[r] + bc;
        if (pe) v += pe[(grow & (SEQ - 1)) * 512 + col];
        if (resid) v += resid[grow * 512 + col];
        const size_t oidx = grow * 512 + col;
        if (outH) outH[oidx] = (_Float16)v;
        if (outF) outF[oidx] = v;
      }
    }
  }
}

// ---------------------------------------------------------------------------
// Fused flash attention + V residual + LayerNorm.
// Block: 256 thr (8 waves), 32 query rows, keys streamed in 128-chunks.
//   score phase: wave w owns keys [w*16,w*16+16); Q/K fragments straight from
//                global (rows are d-contiguous = exact WMMA layout, L2-hot).
//   PV phase:    wave w owns d-cols [w*64,w*64+64); V staged transposed in
//                LDS via 8x8 in-register transpose (b128 in, ds_store_b128).
// Fully masked key chunks are skipped via mask_start.
// ---------------------------------------------------------------------------
__global__ __launch_bounds__(256) void attn_kernel(
    const _Float16* __restrict__ Qh, const _Float16* __restrict__ Kh,
    const _Float16* __restrict__ Vh, const int* __restrict__ mask_start,
    const float* __restrict__ gamma, const float* __restrict__ beta,
    float* __restrict__ Of, _Float16* __restrict__ Ofh) {
  __shared__ __attribute__((aligned(16))) _Float16 Ps[32][136];
  __shared__ __attribute__((aligned(16))) _Float16 Vt[512][40];
  __shared__ float red0[8][32];
  __shared__ float red1[8][32];
  __shared__ float m_s[32], l_s[32], a_s[32], mu_s[32], rs_s[32];

  const int tid = threadIdx.x;
  const int lane = tid & 31, w = tid >> 5;
  const int ln = lane & 15, hi = lane >> 4;
  const int b = blockIdx.y;
  const int q0 = blockIdx.x * 32;
  const int ms = mask_start[b];
  const float scale = 0.044194173824159216f;   // 1/sqrt(512)
  const float NEG_INF = -__builtin_inff();

  const _Float16* Qb = Qh + ((size_t)b * SEQ + q0) * DATT;
  const _Float16* Kb = Kh + (size_t)b * SEQ * DATT;
  const _Float16* Vb = Vh + (size_t)b * SEQ * DATT;

  Acc o[2][4];
#pragma unroll
  for (int mt = 0; mt < 2; ++mt)
#pragma unroll
    for (int nt = 0; nt < 4; ++nt)
#pragma unroll
      for (int r = 0; r < 8; ++r) o[mt][nt].f[r] = 0.0f;

  if (tid < 32) { m_s[tid] = NEG_INF; l_s[tid] = 0.0f; }
  __syncthreads();

  const int nchunks = min(4, (ms + 127) >> 7);
  for (int kc = 0; kc < nchunks; ++kc) {
    // ---- scores: S = Q * K^T ------------------------------------------
    Acc s0, s1;
#pragma unroll
    for (int r = 0; r < 8; ++r) { s0.f[r] = 0.0f; s1.f[r] = 0.0f; }
    const _Float16* Kchunk = Kb + (size_t)(kc * 128 + w * 16) * DATT;
#pragma unroll 4
    for (int dc = 0; dc < 16; ++dc) {
      v16h bf = frag_b(Kchunk + dc * 32, DATT, ln, hi);
      v16h a0 = frag_a(Qb + dc * 32, DATT, ln, hi);
      v16h a1 = frag_a(Qb + 16 * DATT + dc * 32, DATT, ln, hi);
      s0.v = wmma32(a0, bf, s0.v);
      s1.v = wmma32(a1, bf, s1.v);
    }
    // scale + key mask (depends only on this lane's key column)
    const int key = kc * 128 + w * 16 + ln;
    const bool masked = key >= ms;
#pragma unroll
    for (int r = 0; r < 8; ++r) {
      s0.f[r] = masked ? NEG_INF : s0.f[r] * scale;
      s1.f[r] = masked ? NEG_INF : s1.f[r] * scale;
    }
    // per-wave row max -> LDS
#pragma unroll
    for (int mt = 0; mt < 2; ++mt) {
      Acc& s = mt ? s1 : s0;
#pragma unroll
      for (int r = 0; r < 8; ++r) {
        float v = redmax16(s.f[r]);
        if (ln == 0) red0[w][mt * 16 + r + 8 * hi] = v;
      }
    }
    __syncthreads();
    if (tid < 32) {
      float mo = m_s[tid], mn = mo;
#pragma unroll
      for (int i = 0; i < 8; ++i) mn = fmaxf(mn, red0[i][tid]);
      m_s[tid] = mn;
      a_s[tid] = __expf(mo - mn);   // exp(-inf)=0 on first chunk
    }
    __syncthreads();
    // ---- P = exp(S - m), row sums, stage P ----------------------------
#pragma unroll
    for (int mt = 0; mt < 2; ++mt) {
      Acc& s = mt ? s1 : s0;
#pragma unroll
      for (int r = 0; r < 8; ++r) {
        const int row = mt * 16 + r + 8 * hi;
        float p = __expf(s.f[r] - m_s[row]);
        Ps[row][w * 16 + ln] = (_Float16)p;
        float t = redsum16(p);
        if (ln == 0) red0[w][row] = t;
      }
    }
    __syncthreads();
    if (tid < 32) {
      float acc = 0.0f;
#pragma unroll
      for (int i = 0; i < 8; ++i) acc += red0[i][tid];
      l_s[tid] = l_s[tid] * a_s[tid] + acc;
    }
    // ---- rescale running O by alpha -----------------------------------
#pragma unroll
    for (int mt = 0; mt < 2; ++mt)
#pragma unroll
      for (int r = 0; r < 8; ++r) {
        const float a = a_s[mt * 16 + r + 8 * hi];
#pragma unroll
        for (int nt = 0; nt < 4; ++nt) o[mt][nt].f[r] *= a;
      }
    // ---- O += P * V ---------------------------------------------------
    for (int ksub = 0; ksub < 4; ++ksub) {
      __syncthreads();   // protect Vt from previous sub-chunk readers
      {  // stage 32 keys of V transposed via 8x8 register transpose:
         // thread covers keys [k8,k8+8) x d [d0,d0+8); b128 in, b128 out.
        const int k8 = (tid & 3) * 8;
        const int d0 = (tid >> 2) * 8;
        union { uint4 q; _Float16 h[8]; } r[8];
#pragma unroll
        for (int kk = 0; kk < 8; ++kk)
          r[kk].q = *(const uint4*)(
              Vb + (size_t)(kc * 128 + ksub * 32 + k8 + kk) * DATT + d0);
#pragma unroll
        for (int e = 0; e < 8; ++e) {
          union { uint4 q; _Float16 h[8]; } t;
#pragma unroll
          for (int kk = 0; kk < 8; ++kk) t.h[kk] = r[kk].h[e];
          *(uint4*)(&Vt[d0 + e][k8]) = t.q;
        }
      }
      __syncthreads();
      v16h pa0 = frag_a(&Ps[0][ksub * 32], 136, ln, hi);
      v16h pa1 = frag_a(&Ps[16][ksub * 32], 136, ln, hi);
#pragma unroll
      for (int nt = 0; nt < 4; ++nt) {
        v16h vb = frag_b(&Vt[w * 64 + nt * 16][0], 40, ln, hi);
        o[0][nt].v = wmma32(pa0, vb, o[0][nt].v);
        o[1][nt].v = wmma32(pa1, vb, o[1][nt].v);
      }
    }
  }
  __syncthreads();

  // ---- finalize: O/l + V residual, then LayerNorm ---------------------
#pragma unroll
  for (int mt = 0; mt < 2; ++mt) {
#pragma unroll
    for (int r = 0; r < 8; ++r) {
      const int row = mt * 16 + r + 8 * hi;
      const float linv = 1.0f / l_s[row];
      const size_t grow = (size_t)b * SEQ + q0 + row;
      float psum = 0.0f, psq = 0.0f;
#pragma unroll
      for (int nt = 0; nt < 4; ++nt) {
        const int col = w * 64 + nt * 16 + ln;
        float v = o[mt][nt].f[r] * linv + (float)Vh[grow * DATT + col];
        o[mt][nt].f[r] = v;
        psum += v;
        psq += v * v;
      }
      psum = redsum16(psum);
      psq = redsum16(psq);
      if (ln == 0) { red0[w][row] = psum; red1[w][row] = psq; }
    }
  }
  __syncthreads();
  if (tid < 32) {
    float s = 0.0f, q = 0.0f;
#pragma unroll
    for (int i = 0; i < 8; ++i) { s += red0[i][tid]; q += red1[i][tid]; }
    const float mu = s * (1.0f / DATT);
    const float var = q * (1.0f / DATT) - mu * mu;
    mu_s[tid] = mu;
    rs_s[tid] = rsqrtf(var + 1e-5f);
  }
  __syncthreads();
#pragma unroll
  for (int mt = 0; mt < 2; ++mt) {
#pragma unroll
    for (int r = 0; r < 8; ++r) {
      const int row = mt * 16 + r + 8 * hi;
      const float mu = mu_s[row], rs = rs_s[row];
      const size_t grow = (size_t)b * SEQ + q0 + row;
#pragma unroll
      for (int nt = 0; nt < 4; ++nt) {
        const int col = w * 64 + nt * 16 + ln;
        const float v = (o[mt][nt].f[r] - mu) * rs * gamma[col] + beta[col];
        Of[grow * DATT + col] = v;           // exact f32 residual for FFN
        Ofh[grow * DATT + col] = (_Float16)v; // f16 A-operand for FFN GEMM
      }
    }
  }
}

// ---------------------------------------------------------------------------
extern "C" void kernel_launch(void* const* d_in, const int* in_sizes, int n_in,
                              void* d_out, int out_size, void* d_ws,
                              size_t ws_size, hipStream_t stream) {
  (void)in_sizes; (void)n_in; (void)out_size; (void)ws_size;
  const float* x    = (const float*)d_in[0];
  const int* mstart = (const int*)d_in[1];
  const float* Wq = (const float*)d_in[2];
  const float* bq = (const float*)d_in[3];
  const float* Wk = (const float*)d_in[4];
  const float* bk = (const float*)d_in[5];
  const float* Wv = (const float*)d_in[6];
  const float* bv = (const float*)d_in[7];
  const float* Wf = (const float*)d_in[8];
  const float* bf = (const float*)d_in[9];
  const float* pe = (const float*)d_in[10];
  const float* gamma = (const float*)d_in[11];
  const float* beta  = (const float*)d_in[12];

  const size_t Mtot = (size_t)BATCH * SEQ;          // 131072 rows
  _Float16* xh  = (_Float16*)d_ws;                  //  67 MB
  _Float16* Qh  = xh + Mtot * DIN;                  // 134 MB
  _Float16* Kh  = Qh + Mtot * DATT;                 // 134 MB
  _Float16* Vh  = Kh + Mtot * DATT;                 // 134 MB
  _Float16* Ofh = Vh + Mtot * DATT;                 // 134 MB
  _Float16* Wtq = Ofh + Mtot * DATT;                // 512x256 f16
  _Float16* Wtk = Wtq + (size_t)512 * DIN;
  _Float16* Wtv = Wtk + (size_t)512 * DIN;
  _Float16* Wtf = Wtv + (size_t)512 * DIN;          // 512x512 f16
  float* Of = (float*)(Wtf + (size_t)512 * DATT);   // 268 MB  (total ~872 MB)

  dim3 blk(256, 1, 1);

  // 0) conversions / weight transposes (all tiny or pure-bandwidth)
  {
    const long n8 = (long)(Mtot * DIN / 8);
    cvt_f16_kernel<<<dim3((unsigned)((n8 + 255) / 256)), blk, 0, stream>>>(
        x, xh, n8);
    cvt_wt_kernel<<<dim3(16, DIN / 8), blk, 0, stream>>>(Wq, Wtq, DIN);
    cvt_wt_kernel<<<dim3(16, DIN / 8), blk, 0, stream>>>(Wk, Wtk, DIN);
    cvt_wt_kernel<<<dim3(16, DIN / 8), blk, 0, stream>>>(Wv, Wtv, DIN);
    cvt_wt_kernel<<<dim3(16, DATT / 8), blk, 0, stream>>>(Wf, Wtf, DATT);
  }

  dim3 gg(DATT / 128, (unsigned)(Mtot / 128), 1);   // (4, 1024)

  // 1) Q/K/V projections: xh @ W + b + pos_emb -> f16
  gemm_f16g_kernel<<<gg, blk, 0, stream>>>(xh, DIN, Wtq, bq, pe, nullptr,
                                           Qh, nullptr);
  gemm_f16g_kernel<<<gg, blk, 0, stream>>>(xh, DIN, Wtk, bk, pe, nullptr,
                                           Kh, nullptr);
  gemm_f16g_kernel<<<gg, blk, 0, stream>>>(xh, DIN, Wtv, bv, pe, nullptr,
                                           Vh, nullptr);

  // 2) fused masked attention + V residual + LayerNorm -> Of (f32) + Ofh (f16)
  dim3 ga(SEQ / 32, BATCH, 1);                      // (16, 256)
  attn_kernel<<<ga, blk, 0, stream>>>(Qh, Kh, Vh, mstart, gamma, beta, Of,
                                      Ofh);

  // 3) FFN: Ofh @ Wf + bf + Of -> d_out (f32)
  gemm_f16g_kernel<<<gg, blk, 0, stream>>>(Ofh, DATT, Wtf, bf, nullptr, Of,
                                           nullptr, (float*)d_out);
}
